// JitterSeqPredicton_38886633898223
// MI455X (gfx1250) — matrix operand order repaired and checked by reading, not compiled, not run on hardware
//
#include <hip/hip_runtime.h>
#include <hip/hip_bf16.h>
#include <stdint.h>

// ---------------------------------------------------------------------------
// JitterSeqPrediction for MI455X (gfx1250, wave32).
// One 256-thread block (8 wave32s) per row of 2048 int32 tokens.
// Bandwidth-bound: 256 MiB total traffic -> ~11us floor @ 23.3 TB/s.
// The intra-row cumsum (insert/delete scatter destinations) runs on the
// matrix pipe: v_wmma_f32_16x16x32_bf16 with a lower-triangular ones
// A-matrix scans 16 segments x 16 flags per wave (0/1 flags exact in bf16,
// f32 accumulation exact). Segment totals are scanned by wave 0 with a
// shfl-based wave32 scan (1 barrier each side instead of 14 block barriers).
// All LDS traffic on the scan path is b128-vectorized.
// ---------------------------------------------------------------------------

#define EOS_ID   1
#define PAD_ID   0
#define DICT     32000u
#define ROWLEN   2048
#define BLOCK    256
#define PT       8               // elements per thread
#define NSEG     (ROWLEN / 16)   // 128 segments of 16

typedef __attribute__((ext_vector_type(16))) __bf16          v16bf;
typedef __attribute__((ext_vector_type(16))) unsigned short  v16u;
typedef __attribute__((ext_vector_type(8)))  float           v8f;

static __device__ __forceinline__ uint32_t mix32(uint32_t h) {
  h ^= h >> 16; h *= 0x7feb352dU;
  h ^= h >> 15; h *= 0x846ca68bU;
  h ^= h >> 16;
  return h;
}
// Stateless counter-based RNG: (row key, stream id, position) -> u32.
static __device__ __forceinline__ uint32_t hash3(uint32_t key, uint32_t stream, uint32_t pos) {
  return mix32(key ^ mix32(stream * 0x85ebca77U ^ mix32(pos * 0xc2b2ae3dU + 0x27d4eb2fU)));
}
static __device__ __forceinline__ float u01(uint32_t h) {
  return (float)(h >> 8) * (1.0f / 16777216.0f);
}
static __device__ __forceinline__ int rtok(uint32_t h) {
  return (int)(2u + h % (DICT - 2u));   // uniform in [EOS_ID+1, DICT)
}

static __device__ __forceinline__ void store8(int* __restrict__ dst, const int* v) {
  int4 b0 = make_int4(v[0], v[1], v[2], v[3]);
  int4 b1 = make_int4(v[4], v[5], v[6], v[7]);
  ((int4*)dst)[0] = b0;
  ((int4*)dst)[1] = b1;
}

__global__ __launch_bounds__(BLOCK)
void jitter_row_kernel(const int* __restrict__ xs, int* __restrict__ out) {
  __shared__ __align__(32) unsigned short s_flag[ROWLEN];  // bf16 bits of 0/1 flags
  __shared__ __align__(16) int            s_scan[ROWLEN];  // intra-segment incl. prefix
  __shared__ __align__(16) int            s_segtot[NSEG];  // totals -> inclusive scan
  __shared__ __align__(16) int            s_out[ROWLEN];   // scatter staging
  __shared__ int                          s_first;         // first EOS position

  const int row  = blockIdx.x;
  const int tid  = threadIdx.x;
  const int lane = tid & 31;
  const int wave = tid >> 5;
  const int base = tid * PT;

  const uint32_t rk = mix32(0x2a6b17c5u ^ (uint32_t)row * 0x9e3779b9u);

  // ---- load 8 contiguous tokens (32B/thread, 1KB/wave, fully coalesced) ----
  const int4* src = (const int4*)(xs + (size_t)row * ROWLEN + base);
  int4 a0 = src[0];
  int4 a1 = src[1];
  int tok[PT] = {a0.x, a0.y, a0.z, a0.w, a1.x, a1.y, a1.z, a1.w};
  int* gout = out + (size_t)row * ROWLEN + base;

  // ---- first-EOS reduction (ds_min atomic) ----
  if (tid == 0) s_first = ROWLEN;
  __syncthreads();
  int localFirst = ROWLEN;
#pragma unroll
  for (int j = 0; j < PT; ++j)
    if (tok[j] == EOS_ID && localFirst == ROWLEN) localFirst = base + j;
  if (localFirst < ROWLEN) atomicMin(&s_first, localFirst);
  __syncthreads();

  const int first = s_first;
  const int eos   = (first < ROWLEN) ? min(max(first, 1), ROWLEN - 4) : (ROWLEN / 2);

  // Per-row uniform draws (all threads compute identical values).
  const float u = u01(hash3(rk, 0u, 0u));
  const float p = u01(hash3(rk, 1u, 0u));

  if (u > 0.3f) {            // no jitter: stream row through unchanged
    store8(gout, tok);
    return;
  }

  // ---- branch 1: random replace before EOS (pure map, no scan needed) ----
  if (p < 0.33f) {
#pragma unroll
    for (int j = 0; j < PT; ++j) {
      const int i = base + j;
      if (i < eos) {
        const float kp = u01(hash3(rk, 3u, (uint32_t)i));
        if (kp <= 0.2f) tok[j] = rtok(hash3(rk, 2u, (uint32_t)i));
      }
    }
    store8(gout, tok);
    return;
  }

  // ---- branches 2/3 share the flag -> cumsum -> scatter machinery ----
  const bool isInsert = (p < 0.66f);
  const int  maxSel   = (int)((float)(eos + 1) * 0.2f) + 2;   // >= 2
  const uint32_t nStream = isInsert ? 5u : 8u;
  const uint32_t rStream = isInsert ? 4u : 7u;
  const int selN = 1 + (int)(hash3(rk, nStream, 0u) % (uint32_t)(maxSel - 1));
  const float thr = (float)selN / (float)eos;   // Bernoulli stand-in for rank<N

  int sel[PT];
  uint32_t fw[4];            // 8 bf16 flag bit-patterns packed into 4 dwords
#pragma unroll
  for (int k = 0; k < 4; ++k) {
    const int i0 = base + 2 * k;
    const int i1 = i0 + 1;
    sel[2 * k]     = (i0 < eos) && (u01(hash3(rk, rStream, (uint32_t)i0)) < thr);
    sel[2 * k + 1] = (i1 < eos) && (u01(hash3(rk, rStream, (uint32_t)i1)) < thr);
    fw[k] = (sel[2 * k] ? 0x3F80u : 0u) | (sel[2 * k + 1] ? 0x3F800000u : 0u);
  }
  ((uint4*)s_flag)[tid] = make_uint4(fw[0], fw[1], fw[2], fw[3]);  // ds_store_b128
  {
    const int4 z = make_int4(PAD_ID, PAD_ID, PAD_ID, PAD_ID);
    ((int4*)(s_out + base))[0] = z;                                // 2x ds_store_b128
    ((int4*)(s_out + base))[1] = z;
  }
  __syncthreads();

  // ---- WMMA segment scan: one v_wmma per wave scans 256 flags -------------
  // A = 16x16 lower-triangular ones embedded in 16x32 bf16 (K>=16 zero).
//     Lanes 0-15 carry K=0..15; lanes 16-31 carry K=16..31 = zero.
  // B = 16 segments of 16 flags: lanes 0-15 carry K=0..15 of column N=lane
  //     (2x ds_load_b128); lanes 16-31 carry K=16..31 = zero padding.
  // D[m][n] = inclusive prefix of segment n at position m (exact in f32).
  {
    const int M    = lane & 15;
    const int kofs = (lane < 16) ? 0 : 16;
    v16u au = {};
#pragma unroll
    for (int j = 0; j < 16; ++j)
      au[j] = ((j + kofs) <= M) ? (unsigned short)0x3F80u : (unsigned short)0u;

    v16u bu = {};
    if (lane < 16)
      bu = ((const v16u*)s_flag)[wave * 16 + M];   // 32B aligned -> 2x ds_load_b128

    const v16bf A = __builtin_bit_cast(v16bf, au);
    const v16bf B = __builtin_bit_cast(v16bf, bu);
    v8f C = {};
    v8f D = __builtin_amdgcn_wmma_f32_16x16x32_bf16(
        /*neg_a=*/false, A, /*neg_b=*/false, B,
        /*c_mod=*/(short)0, C, /*reuse_a=*/false, /*reuse_b=*/false);

    const int n  = lane & 15;
    const int mb = (lane < 16) ? 0 : 8;
    int pr[8];
#pragma unroll
    for (int r = 0; r < 8; ++r) pr[r] = (int)D[r];
    store8(&s_scan[wave * 256 + n * 16 + mb], pr);   // 2x ds_store_b128
    if (lane >= 16) s_segtot[wave * 16 + n] = pr[7]; // segment total (M=15)
  }
  __syncthreads();

  // ---- wave 0: inclusive scan of 128 segment totals (shfl wave scan) ------
  if (wave == 0) {
    int4 t = ((const int4*)s_segtot)[lane];          // 4 totals per lane (b128)
    t.y += t.x; t.z += t.y; t.w += t.z;              // serial scan of 4
    int s = t.w;
#pragma unroll
    for (int off = 1; off < 32; off <<= 1) {         // wave32 inclusive scan
      const int v = __shfl_up(s, off, 32);
      if (lane >= off) s += v;
    }
    const int excl = s - t.w;                        // exclusive lane offset
    t.x += excl; t.y += excl; t.z += excl; t.w += excl;
    ((int4*)s_segtot)[lane] = t;
  }
  __syncthreads();

  // ---- scatter into LDS staging using cumsum destinations -----------------
  if (isInsert) {
#pragma unroll
    for (int j = 0; j < PT; ++j) {
      const int i   = base + j;
      const int seg = i >> 4;
      const int ci  = ((seg > 0) ? s_segtot[seg - 1] : 0) + s_scan[i];  // inclusive
      if (i <= eos) {
        const int d = i + ci;                    // orig_dest = pos + c_i
        if (d < ROWLEN) s_out[d] = tok[j];
      }
      if (sel[j]) {
        const int d2 = i + ci - 1;               // ins_dest = pos + c_i - 1
        if (d2 >= 0 && d2 < ROWLEN) {
          const int vi = min(max(ci - 1, 0), ROWLEN - 1);
          s_out[d2] = rtok(hash3(rk, 6u, (uint32_t)vi));
        }
      }
    }
    __syncthreads();
    const bool revert = (eos + 1 + selN > ROWLEN);   // grown seq exceeds length
    if (revert) {
      store8(gout, tok);
    } else {
      int o[PT];
#pragma unroll
      for (int j = 0; j < PT; ++j) o[j] = s_out[base + j];
      store8(gout, o);
    }
  } else {  // delete
#pragma unroll
    for (int j = 0; j < PT; ++j) {
      const int i   = base + j;
      const int seg = i >> 4;
      const int cd  = ((seg > 0) ? s_segtot[seg - 1] : 0) + s_scan[i];
      const bool keep = (i <= eos) && !sel[j];
      if (keep) {
        const int d = i - cd;                    // del_dest = pos - c_d (cd==excl here)
        if (d >= 0) s_out[d] = tok[j];
      }
    }
    __syncthreads();
    int o[PT];
#pragma unroll
    for (int j = 0; j < PT; ++j) o[j] = s_out[base + j];
    store8(gout, o);
  }
}

extern "C" void kernel_launch(void* const* d_in, const int* in_sizes, int n_in,
                              void* d_out, int out_size, void* d_ws, size_t ws_size,
                              hipStream_t stream) {
  const int* xs = (const int*)d_in[0];
  int* out      = (int*)d_out;          // int32 tokens (4B elements)
  const int rows = in_sizes[0] / ROWLEN;
  jitter_row_kernel<<<dim3(rows), dim3(BLOCK), 0, stream>>>(xs, out);
}
